// rel_multihead_attn_56908316672217
// MI455X (gfx1250) — compile-verified, matrix operand order checked by reading, not run on hardware
//
#include <hip/hip_runtime.h>
#include <hip/hip_bf16.h>

typedef __attribute__((ext_vector_type(16))) _Float16 v16h;
typedef __attribute__((ext_vector_type(8)))  _Float16 v8h;
typedef __attribute__((ext_vector_type(8)))  float    v8f;

#define QLEN  1024
#define EMB   1024
#define NHEAD 16
#define DHEAD 64
#define BATCH 4

#if defined(__gfx1250__) && __has_builtin(__builtin_amdgcn_tensor_load_to_lds)
#define USE_TDM 1
#else
#define USE_TDM 0
#endif

// ---------------------------------------------------------------------------
// WMMA helpers (CDNA5 gfx1250, wave32)
// ---------------------------------------------------------------------------
__device__ __forceinline__ v8f wmma_f16(v16h a, v16h b, v8f c) {
  // v_wmma_f32_16x16x32_f16  D = A(16x32) x B(32x16) + C
  return __builtin_amdgcn_wmma_f32_16x16x32_f16(false, a, false, b, (short)0, c,
                                                false, false);
}

// A-fragment (16x32 f16): lane L<16 holds row L, K {0..7}U{16..23};
// lane L>=16 holds row L-16, K {8..15}U{24..31}.
// Identical pattern serves B-fragments when B is supplied transposed
// (rows of `base` = columns of B, contiguous along K).
__device__ __forceinline__ v16h load_frag(const _Float16* base, int row0,
                                          int ld, int k0) {
  int lane = threadIdx.x & 31;
  int r = lane & 15, h = lane >> 4;
  const _Float16* p = base + (size_t)(row0 + r) * ld + k0 + h * 8;
  v16h f;
  *(v8h*)&f       = *(const v8h*)(p);
  *((v8h*)&f + 1) = *(const v8h*)(p + 16);
  return f;
}

__device__ __forceinline__ v16h load_frag_clamp(const _Float16* base, int row0,
                                                int rmax, int ld, int k0) {
  int lane = threadIdx.x & 31;
  int r = lane & 15, h = lane >> 4;
  int rr = row0 + r;
  rr = rr < 0 ? 0 : (rr > rmax ? rmax : rr);
  const _Float16* p = base + (size_t)rr * ld + k0 + h * 8;
  v16h f;
  *(v8h*)&f       = *(const v8h*)(p);
  *((v8h*)&f + 1) = *(const v8h*)(p + 16);
  return f;
}

#if USE_TDM
// ---------------------------------------------------------------------------
// Tensor Data Mover: async 2D f16 tile (tile_w x tile_h, row stride
// stride_elems) from global -> contiguous LDS.  D# per CDNA5 ISA ch.8.
// ---------------------------------------------------------------------------
typedef __attribute__((ext_vector_type(4))) unsigned int tdm_v4u;
typedef __attribute__((ext_vector_type(8))) int          tdm_v8i;
typedef __attribute__((ext_vector_type(4))) int          tdm_v4i;

__device__ __forceinline__ void tdm_load_2d(unsigned lds_off,
                                            const _Float16* gptr,
                                            unsigned tile_w, unsigned tile_h,
                                            unsigned stride_elems) {
  unsigned long long ga = (unsigned long long)(uintptr_t)gptr;
  tdm_v4u g0;
  g0[0] = 1u;                                   // count=1 (valid user D#)
  g0[1] = lds_off;                              // lds_addr
  g0[2] = (unsigned)(ga & 0xffffffffu);         // global_addr[31:0]
  g0[3] = (unsigned)((ga >> 32) & 0x01ffffffu)  // global_addr[56:32]
          | 0x80000000u;                        // type=2 ("image")
  tdm_v8i g1;
  g1[0] = (int)(1u << 16);                      // data_size=1 (2B), mask=0
  g1[1] = (int)((tile_w & 0xffffu) << 16);      // tensor_dim0[15:0]
  g1[2] = (int)((tile_w >> 16) | ((tile_h & 0xffffu) << 16));  // td0hi|td1lo
  g1[3] = (int)((tile_h >> 16) | ((tile_w & 0xffffu) << 16));  // td1hi|tile0
  g1[4] = (int)(tile_h & 0xffffu);              // tile_dim1 (tile_dim2=0)
  g1[5] = (int)stride_elems;                    // tensor_dim0_stride[31:0]
  g1[6] = 0;
  g1[7] = 0;
  tdm_v4i z4 = {0, 0, 0, 0};
#if __clang_major__ >= 23
  tdm_v8i z8 = {0, 0, 0, 0, 0, 0, 0, 0};
  __builtin_amdgcn_tensor_load_to_lds(g0, g1, z4, z4, z8, 0);
#else
  __builtin_amdgcn_tensor_load_to_lds(g0, g1, z4, z4, 0);
#endif
}
#endif  // USE_TDM

// ---------------------------------------------------------------------------
// f32 -> f16 conversion / weight transpose
// ---------------------------------------------------------------------------
__global__ void __launch_bounds__(256) cvt_f32_to_f16(const float* __restrict__ src,
                                                      _Float16* __restrict__ dst,
                                                      int nelem) {
  int i = blockIdx.x * 256 + threadIdx.x;
  if (i < nelem) dst[i] = (_Float16)src[i];
}

// WT[n*K + k] = W[k*N + n]   (so B-fragment K runs contiguously)
__global__ void __launch_bounds__(256) transpose_f32_to_f16(const float* __restrict__ W,
                                                            _Float16* __restrict__ WT,
                                                            int K, int N) {
  int i = blockIdx.x * 256 + threadIdx.x;
  if (i < K * N) {
    int nn = i / K, kk = i % K;
    WT[i] = (_Float16)W[(size_t)kk * N + nn];
  }
}

// ---------------------------------------------------------------------------
// f16 WMMA GEMM: C = A(MxK) * BT^T(KxN) [+ bias(N)]
// 8 waves/block; wave = 16x64 strip; block = 128x64.
// B tile (64x32) is identical for all 8 waves -> staged once per k-step by
// TDM into a double-buffered LDS stage (one DMA always in flight); A frags
// register double-buffered; all 4 B frags loaded before the 4 WMMAs so the
// scheduler can use partial dscnt waits.  mode 0/1/2 = f16 / f32 / vT out.
// ---------------------------------------------------------------------------
__global__ void __launch_bounds__(256) gemm_f16_kernel(
    const _Float16* __restrict__ A, const _Float16* __restrict__ BT,
    const float* __restrict__ bias, _Float16* __restrict__ outH,
    float* __restrict__ outF, int M, int N, int K, int mode) {
  int wv = threadIdx.x >> 5;
  int lane = threadIdx.x & 31;
  int row0 = blockIdx.y * 128 + wv * 16;
  int col0 = blockIdx.x * 64;
  const int KSTEPS = K / 32;

  v8f acc[4] = {v8f{}, v8f{}, v8f{}, v8f{}};

#if USE_TDM
  __shared__ _Float16 bstage[2][64 * 32];  // 2 x 4KB double buffer
  if (wv == 0)
    tdm_load_2d((unsigned)(uintptr_t)&bstage[0][0], BT + (size_t)col0 * K,
                32, 64, (unsigned)K);
  v16h a_cur = load_frag(A, row0, K, 0);
  for (int s = 0; s < KSTEPS; ++s) {
    // prefetch next A fragment (global) before the barrier so it overlaps
    v16h a_nxt = a_cur;
    if (s + 1 < KSTEPS) a_nxt = load_frag(A, row0, K, (s + 1) * 32);
    if (wv == 0) {
      if (s + 1 < KSTEPS) {
        tdm_load_2d((unsigned)(uintptr_t)&bstage[(s + 1) & 1][0],
                    BT + (size_t)col0 * K + (s + 1) * 32, 32, 64, (unsigned)K);
        __builtin_amdgcn_s_wait_tensorcnt(1);  // current buffer done
      } else {
        __builtin_amdgcn_s_wait_tensorcnt(0);
      }
    }
    __syncthreads();
    const _Float16* bs = &bstage[s & 1][0];
    v16h bfr[4];
#pragma unroll
    for (int nt = 0; nt < 4; ++nt) bfr[nt] = load_frag(bs, nt * 16, 32, 0);
#pragma unroll
    for (int nt = 0; nt < 4; ++nt) acc[nt] = wmma_f16(a_cur, bfr[nt], acc[nt]);
    a_cur = a_nxt;
    __syncthreads();  // reads of bstage[s&1] done before step s+1 refills it
  }
#else
  // Fallback: register double-buffered global fragments.
  v16h a_cur = load_frag(A, row0, K, 0);
  v16h b_cur[4];
#pragma unroll
  for (int nt = 0; nt < 4; ++nt) b_cur[nt] = load_frag(BT, col0 + nt * 16, K, 0);
  for (int s = 0; s < KSTEPS; ++s) {
    v16h a_nxt = a_cur;
    v16h b_nxt[4];
    if (s + 1 < KSTEPS) {
      int kn = (s + 1) * 32;
      a_nxt = load_frag(A, row0, K, kn);
#pragma unroll
      for (int nt = 0; nt < 4; ++nt) b_nxt[nt] = load_frag(BT, col0 + nt * 16, K, kn);
    } else {
#pragma unroll
      for (int nt = 0; nt < 4; ++nt) b_nxt[nt] = b_cur[nt];
    }
#pragma unroll
    for (int nt = 0; nt < 4; ++nt) acc[nt] = wmma_f16(a_cur, b_cur[nt], acc[nt]);
    a_cur = a_nxt;
#pragma unroll
    for (int nt = 0; nt < 4; ++nt) b_cur[nt] = b_nxt[nt];
  }
#endif

  int n = lane & 15, hs = lane >> 4;
#pragma unroll
  for (int nt = 0; nt < 4; ++nt) {
#pragma unroll
    for (int v = 0; v < 8; ++v) {
      int m = v + 8 * hs;  // C/D layout: lane=col, vgpr=row
      int grow = row0 + m;
      int gcol = col0 + nt * 16 + n;
      float val = acc[nt][v];
      if (bias) val += bias[gcol];
      if (mode == 0) {
        outH[(size_t)grow * N + gcol] = (_Float16)val;
      } else if (mode == 1) {
        outF[(size_t)grow * N + gcol] = val;
      } else {  // V stored (b, head, d, q)
        int bb = grow / QLEN, q = grow % QLEN;
        int head = gcol >> 6, d = gcol & 63;
        outH[(((size_t)bb * NHEAD + head) * DHEAD + d) * QLEN + q] = (_Float16)val;
      }
    }
  }
}

// ---------------------------------------------------------------------------
// Fused flash attention with Transformer-XL relative shift.
// rel_shift => BD[i,j] = qr[i] . rp[Q-1-i+j]  (only j<=i survives the mask)
// Block: (b,head) x 64-query tile; 4 waves x 16 rows; stream 64-key blocks.
// K-tile and V-tile of each key block staged once by TDM into LDS.
// Fragment loads batched per k-chunk ahead of the WMMA chains.
// ---------------------------------------------------------------------------
__global__ void __launch_bounds__(128) attn_kernel(
    const _Float16* __restrict__ qw, const _Float16* __restrict__ qr,
    const _Float16* __restrict__ kmat, const _Float16* __restrict__ vT,
    const _Float16* __restrict__ rp, _Float16* __restrict__ o16) {
  __shared__ float    bdw_lds[4][16][80];  // per-wave 16x80 BD band tile
  __shared__ _Float16 p_lds[4][16][64];    // per-wave P re-fragment buffer
#if USE_TDM
  __shared__ _Float16 k_lds[64][64];       // key block (j, d) slice
  __shared__ _Float16 v_lds[64][64];       // value block (d, j) slice
#endif

  int bh = blockIdx.y;  // b*16 + head
  int b = bh >> 4;
  int i0blk = blockIdx.x * 64;
  int wv = threadIdx.x >> 5;
  int lane = threadIdx.x & 31;
  int i0w = i0blk + wv * 16;
  int kcol = (bh & 15) * DHEAD;
  const _Float16* qw_b = qw + (size_t)b * QLEN * EMB;
  const _Float16* qr_b = qr + (size_t)b * QLEN * EMB;
  const _Float16* k_b  = kmat + (size_t)b * QLEN * EMB;
  const _Float16* vT_h = vT + (size_t)bh * DHEAD * QLEN;

  v16h qwA[2], qrA[2];
#pragma unroll
  for (int kc = 0; kc < 2; ++kc) {
    qwA[kc] = load_frag(qw_b, i0w, EMB, kcol + kc * 32);
    qrA[kc] = load_frag(qr_b, i0w, EMB, kcol + kc * 32);
  }
  float mrow[8], lrow[8];
  v8f oacc[4] = {v8f{}, v8f{}, v8f{}, v8f{}};
#pragma unroll
  for (int v = 0; v < 8; ++v) { mrow[v] = -3.0e38f; lrow[v] = 0.f; }
  int n = lane & 15, hs = lane >> 4;

  for (int j0 = 0; j0 <= i0blk; j0 += 64) {
#if USE_TDM
    if (wv == 0) {
      tdm_load_2d((unsigned)(uintptr_t)&k_lds[0][0],
                  k_b + (size_t)j0 * EMB + kcol, 64, 64, EMB);
      tdm_load_2d((unsigned)(uintptr_t)&v_lds[0][0], vT_h + j0, 64, 64, QLEN);
      __builtin_amdgcn_s_wait_tensorcnt(0);
    }
    __syncthreads();
#endif
    // ---- AC = (q + r_w_bias) . k^T : batch 4 frags per k-chunk ----
    v8f s[4];
    {
      v16h bk[4];
#pragma unroll
      for (int nt = 0; nt < 4; ++nt)
#if USE_TDM
        bk[nt] = load_frag(&k_lds[0][0], nt * 16, 64, 0);
#else
        bk[nt] = load_frag(k_b, j0 + nt * 16, EMB, kcol);
#endif
#pragma unroll
      for (int nt = 0; nt < 4; ++nt) {
        v8f c = {};
        s[nt] = wmma_f16(qwA[0], bk[nt], c);
      }
#pragma unroll
      for (int nt = 0; nt < 4; ++nt)
#if USE_TDM
        bk[nt] = load_frag(&k_lds[0][0], nt * 16, 64, 32);
#else
        bk[nt] = load_frag(k_b, j0 + nt * 16, EMB, kcol + 32);
#endif
#pragma unroll
      for (int nt = 0; nt < 4; ++nt) s[nt] = wmma_f16(qwA[1], bk[nt], s[nt]);
    }
    // ---- BD band: 16x80 window of rp, batched, then shifted gather ----
    int lbase = (QLEN - 16) - i0w + j0;  // >= 0 always
    {
      v16h br[5];
      v8f bd[5];
#pragma unroll
      for (int nt = 0; nt < 5; ++nt)
        br[nt] = load_frag_clamp(rp, lbase + nt * 16, QLEN - 1, EMB, kcol);
#pragma unroll
      for (int nt = 0; nt < 5; ++nt) {
        v8f c = {};
        bd[nt] = wmma_f16(qrA[0], br[nt], c);
      }
#pragma unroll
      for (int nt = 0; nt < 5; ++nt)
        br[nt] = load_frag_clamp(rp, lbase + nt * 16, QLEN - 1, EMB, kcol + 32);
#pragma unroll
      for (int nt = 0; nt < 5; ++nt) bd[nt] = wmma_f16(qrA[1], br[nt], bd[nt]);
#pragma unroll
      for (int nt = 0; nt < 5; ++nt)
#pragma unroll
        for (int v = 0; v < 8; ++v)
          bdw_lds[wv][v + 8 * hs][nt * 16 + n] = bd[nt][v];
    }
    __syncthreads();

    // ---- assemble scores, causal mask, online softmax ----
    float rmax[8];
#pragma unroll
    for (int v = 0; v < 8; ++v) rmax[v] = -3.0e38f;
#pragma unroll
    for (int nt = 0; nt < 4; ++nt) {
#pragma unroll
      for (int v = 0; v < 8; ++v) {
        int row = v + 8 * hs;
        int jloc = nt * 16 + n;
        float val = s[nt][v] + bdw_lds[wv][row][15 - row + jloc];
        val *= 0.03125f;                            // 1/sqrt(1024)
        if (j0 + jloc > i0w + row) val = -1.0e30f;  // mask j > i
        s[nt][v] = val;
        rmax[v] = fmaxf(rmax[v], val);
      }
    }
#pragma unroll
    for (int v = 0; v < 8; ++v)
#pragma unroll
      for (int st = 1; st < 16; st <<= 1)
        rmax[v] = fmaxf(rmax[v], __shfl_xor(rmax[v], st, 32));
    float scale[8], rsum[8];
#pragma unroll
    for (int v = 0; v < 8; ++v) {
      float mnew = fmaxf(mrow[v], rmax[v]);
      scale[v] = __expf(mrow[v] - mnew);
      mrow[v] = mnew;
      rsum[v] = 0.f;
    }
#pragma unroll
    for (int nt = 0; nt < 4; ++nt)
#pragma unroll
      for (int v = 0; v < 8; ++v) {
        float p = __expf(s[nt][v] - mrow[v]);
        s[nt][v] = p;
        rsum[v] += p;
      }
#pragma unroll
    for (int v = 0; v < 8; ++v) {
#pragma unroll
      for (int st = 1; st < 16; st <<= 1) rsum[v] += __shfl_xor(rsum[v], st, 32);
      lrow[v] = lrow[v] * scale[v] + rsum[v];
    }
#pragma unroll
    for (int nt = 0; nt < 4; ++nt)
#pragma unroll
      for (int v = 0; v < 8; ++v) oacc[nt][v] *= scale[v];

    // ---- stage P (D-layout -> A-layout via LDS), then O += P.V ----
#pragma unroll
    for (int nt = 0; nt < 4; ++nt)
#pragma unroll
      for (int v = 0; v < 8; ++v)
        p_lds[wv][v + 8 * hs][nt * 16 + n] = (_Float16)s[nt][v];
    __syncthreads();
#pragma unroll
    for (int kc = 0; kc < 2; ++kc) {
      v16h pa = load_frag(&p_lds[wv][0][0], 0, DHEAD, kc * 32);
      v16h bv[4];
#pragma unroll
      for (int nt = 0; nt < 4; ++nt)
#if USE_TDM
        bv[nt] = load_frag(&v_lds[0][0], nt * 16, 64, kc * 32);
#else
        bv[nt] = load_frag(vT_h, nt * 16, QLEN, j0 + kc * 32);
#endif
#pragma unroll
      for (int nt = 0; nt < 4; ++nt) oacc[nt] = wmma_f16(pa, bv[nt], oacc[nt]);
    }
#if USE_TDM
    __syncthreads();  // k_lds/v_lds reads done before next block's DMA
#endif
  }
  // ---- finalize: divide by row sum, store f16 ----
#pragma unroll
  for (int nt = 0; nt < 4; ++nt)
#pragma unroll
    for (int v = 0; v < 8; ++v) {
      int row = v + 8 * hs;
      o16[((size_t)b * QLEN + i0w + row) * EMB + kcol + nt * 16 + n] =
          (_Float16)(oacc[nt][v] / lrow[v]);
    }
}

// ---------------------------------------------------------------------------
// Residual + LayerNorm: out = LN(w + attn_out) * gamma + beta
// ---------------------------------------------------------------------------
__global__ void __launch_bounds__(256) ln_kernel(const float* __restrict__ w,
                                                 const float* __restrict__ t,
                                                 const float* __restrict__ gamma,
                                                 const float* __restrict__ beta,
                                                 float* __restrict__ out) {
  int row = blockIdx.x;
  const float* wr = w + (size_t)row * EMB;
  const float* tr = t + (size_t)row * EMB;
  float* orow = out + (size_t)row * EMB;
  float vals[4], s1 = 0.f, s2 = 0.f;
#pragma unroll
  for (int i = 0; i < 4; ++i) {
    int c = threadIdx.x + i * 256;
    float v = wr[c] + tr[c];
    vals[i] = v;
    s1 += v;
    s2 += v * v;
  }
#pragma unroll
  for (int st = 1; st < 32; st <<= 1) {
    s1 += __shfl_xor(s1, st, 32);
    s2 += __shfl_xor(s2, st, 32);
  }
  __shared__ float a1[8], a2[8];
  int wv = threadIdx.x >> 5, lane = threadIdx.x & 31;
  if (lane == 0) { a1[wv] = s1; a2[wv] = s2; }
  __syncthreads();
  float t1 = 0.f, t2 = 0.f;
#pragma unroll
  for (int j = 0; j < 8; ++j) { t1 += a1[j]; t2 += a2[j]; }
  float mu = t1 * (1.0f / EMB);
  float var = t2 * (1.0f / EMB) - mu * mu;
  float inv = rsqrtf(var + 1.0e-3f);
#pragma unroll
  for (int i = 0; i < 4; ++i) {
    int c = threadIdx.x + i * 256;
    orow[c] = (vals[i] - mu) * inv * gamma[c] + beta[c];
  }
}

// ---------------------------------------------------------------------------
extern "C" void kernel_launch(void* const* d_in, const int* in_sizes, int n_in,
                              void* d_out, int out_size, void* d_ws,
                              size_t ws_size, hipStream_t stream) {
  const float* w     = (const float*)d_in[0];
  const float* r     = (const float*)d_in[1];
  const float* rwb   = (const float*)d_in[2];
  const float* rrb   = (const float*)d_in[3];
  // d_in[4] attn_mask: pure causal, applied analytically in attn_kernel
  const float* Wq    = (const float*)d_in[5];
  const float* Wk    = (const float*)d_in[6];
  const float* Wv    = (const float*)d_in[7];
  const float* Wr    = (const float*)d_in[8];
  const float* Wo    = (const float*)d_in[9];
  const float* gamma = (const float*)d_in[10];
  const float* beta  = (const float*)d_in[11];
  float* out = (float*)d_out;

  char* ws = (char*)d_ws;
  size_t off = 0;
  auto alloc = [&](size_t bytes) {
    char* p = ws + off;
    off += (bytes + 255) & ~(size_t)255;
    return p;
  };
  const int M = BATCH * QLEN;  // 4096
  _Float16* w16  = (_Float16*)alloc((size_t)M * EMB * 2);
  _Float16* r16  = (_Float16*)alloc((size_t)QLEN * EMB * 2);
  _Float16* WqT  = (_Float16*)alloc((size_t)EMB * EMB * 2);
  _Float16* WkT  = (_Float16*)alloc((size_t)EMB * EMB * 2);
  _Float16* WvT  = (_Float16*)alloc((size_t)EMB * EMB * 2);
  _Float16* WrT  = (_Float16*)alloc((size_t)EMB * EMB * 2);
  _Float16* WoT  = (_Float16*)alloc((size_t)EMB * EMB * 2);
  _Float16* qw16 = (_Float16*)alloc((size_t)M * EMB * 2);
  _Float16* qr16 = (_Float16*)alloc((size_t)M * EMB * 2);
  _Float16* k16  = (_Float16*)alloc((size_t)M * EMB * 2);
  _Float16* vT16 = (_Float16*)alloc((size_t)M * EMB * 2);
  _Float16* rp16 = (_Float16*)alloc((size_t)QLEN * EMB * 2);
  _Float16* o16  = (_Float16*)alloc((size_t)M * EMB * 2);
  float*    tmp  = (float*)alloc((size_t)M * EMB * 4);

  // 1) precision conversion + weight transposes
  cvt_f32_to_f16<<<M * EMB / 256, 256, 0, stream>>>(w, w16, M * EMB);
  cvt_f32_to_f16<<<QLEN * EMB / 256, 256, 0, stream>>>(r, r16, QLEN * EMB);
  transpose_f32_to_f16<<<EMB * EMB / 256, 256, 0, stream>>>(Wq, WqT, EMB, EMB);
  transpose_f32_to_f16<<<EMB * EMB / 256, 256, 0, stream>>>(Wk, WkT, EMB, EMB);
  transpose_f32_to_f16<<<EMB * EMB / 256, 256, 0, stream>>>(Wv, WvT, EMB, EMB);
  transpose_f32_to_f16<<<EMB * EMB / 256, 256, 0, stream>>>(Wr, WrT, EMB, EMB);
  transpose_f32_to_f16<<<EMB * EMB / 256, 256, 0, stream>>>(Wo, WoT, EMB, EMB);

  // 2) projection GEMMs (WMMA f16 -> f32 acc, TDM-staged B tiles)
  dim3 gA(EMB / 64, M / 128);
  gemm_f16_kernel<<<gA, 256, 0, stream>>>(w16, WqT, rwb, qw16, nullptr, M, EMB, EMB, 0);
  gemm_f16_kernel<<<gA, 256, 0, stream>>>(w16, WqT, rrb, qr16, nullptr, M, EMB, EMB, 0);
  gemm_f16_kernel<<<gA, 256, 0, stream>>>(w16, WkT, nullptr, k16, nullptr, M, EMB, EMB, 0);
  gemm_f16_kernel<<<gA, 256, 0, stream>>>(w16, WvT, nullptr, vT16, nullptr, M, EMB, EMB, 2);
  dim3 gR(EMB / 64, QLEN / 128);
  gemm_f16_kernel<<<gR, 256, 0, stream>>>(r16, WrT, nullptr, rp16, nullptr, QLEN, EMB, EMB, 0);

  // 3) fused flash attention with rel-shift (TDM-staged K/V tiles)
  dim3 gAttn(QLEN / 64, BATCH * NHEAD);
  attn_kernel<<<gAttn, 128, 0, stream>>>(qw16, qr16, k16, vT16, rp16, o16);

  // 4) output projection + residual LayerNorm
  gemm_f16_kernel<<<gA, 256, 0, stream>>>(o16, WoT, nullptr, nullptr, tmp, M, EMB, EMB, 1);
  ln_kernel<<<M, 256, 0, stream>>>(w, tmp, gamma, beta, out);

  (void)n_in; (void)in_sizes; (void)out_size; (void)ws_size;
}
